// VariableSelectionNetwork_61632780697951
// MI455X (gfx1250) — matrix operand order repaired and checked by reading, not compiled, not run on hardware
//
#include <hip/hip_runtime.h>
#include <hip/hip_bf16.h>
#include <math.h>

#define B_   256
#define T_   128
#define F_   4
#define H_   512
#define BT_  (B_*T_)      /* 32768 rows */
#define MT_  64           /* rows per workgroup in the fused main kernel */
#define EPS_ 1e-3f

typedef __attribute__((ext_vector_type(16))) __bf16 v16bf;
typedef __attribute__((ext_vector_type(8)))  float  v8f;

union Frag16 { uint4 u[2]; v16bf v; };

__device__ __forceinline__ unsigned short f2bf(float f) {
  unsigned int u = __float_as_uint(f);
  unsigned int r = 0x7FFFu + ((u >> 16) & 1u);   // round-to-nearest-even
  return (unsigned short)((u + r) >> 16);
}
__device__ __forceinline__ float bf2f(unsigned short b) {
  return __uint_as_float(((unsigned int)b) << 16);
}
__device__ __forceinline__ float eluf(float x)     { return x > 0.f ? x : (__expf(x) - 1.f); }
__device__ __forceinline__ float sigmoidf_(float x){ return 1.f / (1.f + __expf(-x)); }

// ---------------------------------------------------------------------------
// Kernel 1: convert W2/Wg [F,H,H] f32 (k-major) -> bf16 transposed [F,H(n),H(k)]
// so WMMA B-fragments are K-contiguous per output column.
// ---------------------------------------------------------------------------
__global__ __launch_bounds__(256) void vsn_prep_kernel(
    const float* __restrict__ W2, const float* __restrict__ Wg,
    unsigned short* __restrict__ W2t, unsigned short* __restrict__ Wgt)
{
  unsigned int i = blockIdx.x * 256u + threadIdx.x;   // exactly F_*H_*H_ threads
  unsigned int k = i & (H_ - 1);
  unsigned int n = (i >> 9) & (H_ - 1);
  unsigned int f = i >> 18;
  size_t src = ((size_t)f * H_ + k) * H_ + n;
  W2t[i] = f2bf(W2[src]);
  Wgt[i] = f2bf(Wg[src]);
}

// ---------------------------------------------------------------------------
// Kernel 2: context GRN + softmax selection (tiny: 0.4 GFLOP fp32 VALU).
// One workgroup per batch row b. Writes sel[B,F] to ws and sel_weights output.
// ---------------------------------------------------------------------------
__global__ __launch_bounds__(256) void vsn_context_kernel(
    const float* __restrict__ inputs,
    const float* __restrict__ W1c, const float* __restrict__ b1c,
    const float* __restrict__ W2c, const float* __restrict__ b2c,
    const float* __restrict__ Wgc, const float* __restrict__ bgc,
    const float* __restrict__ gamma_c, const float* __restrict__ beta_c,
    const float* __restrict__ Wsel, const float* __restrict__ bsel,
    float* __restrict__ selOut, float* __restrict__ selWOut)
{
  __shared__ float sFlat[H_], sA[H_], sBuf[H_];
  __shared__ float sRed[256], sRed2[256];
  __shared__ float sPart[256 * F_];
  __shared__ float sSel4[F_];
  const int tid = threadIdx.x;
  const int b = blockIdx.x;

  for (int i = tid; i < H_; i += 256) sFlat[i] = inputs[(size_t)b * H_ + i];
  __syncthreads();

  // x = elu(flat @ W1c + b1c)
  for (int n = tid; n < H_; n += 256) {
    float acc = b1c[n];
    for (int k = 0; k < H_; ++k) acc = fmaf(sFlat[k], W1c[(size_t)k * H_ + n], acc);
    sA[n] = eluf(acc);
  }
  __syncthreads();

  // x2 = x @ W2c + b2c
  for (int n = tid; n < H_; n += 256) {
    float acc = b2c[n];
    for (int k = 0; k < H_; ++k) acc = fmaf(sA[k], W2c[(size_t)k * H_ + n], acc);
    sBuf[n] = acc;
  }
  __syncthreads();

  // g = sigmoid(x2 @ Wgc + bgc); h = g*x2 + (1-g)*flat  -> sA
  for (int n = tid; n < H_; n += 256) {
    float acc = bgc[n];
    for (int k = 0; k < H_; ++k) acc = fmaf(sBuf[k], Wgc[(size_t)k * H_ + n], acc);
    float g = sigmoidf_(acc);
    sA[n] = g * sBuf[n] + (1.f - g) * sFlat[n];
  }
  __syncthreads();

  // LayerNorm(h) -> ctx (sBuf)
  float p1 = 0.f, p2 = 0.f;
  for (int i = tid; i < H_; i += 256) { float x = sA[i]; p1 += x; p2 += x * x; }
  sRed[tid] = p1; sRed2[tid] = p2;
  __syncthreads();
  for (int s = 128; s > 0; s >>= 1) {
    if (tid < s) { sRed[tid] += sRed[tid + s]; sRed2[tid] += sRed2[tid + s]; }
    __syncthreads();
  }
  float m    = sRed[0]  * (1.f / H_);
  float var  = sRed2[0] * (1.f / H_) - m * m;
  float rstd = rsqrtf(var + EPS_);
  for (int n = tid; n < H_; n += 256)
    sBuf[n] = gamma_c[n] * (sA[n] - m) * rstd + beta_c[n];
  __syncthreads();

  // logits = ctx @ Wsel + bsel ; softmax
  float part[F_];
#pragma unroll
  for (int j = 0; j < F_; ++j) part[j] = 0.f;
  for (int k = tid; k < H_; k += 256)
#pragma unroll
    for (int j = 0; j < F_; ++j) part[j] = fmaf(sBuf[k], Wsel[k * F_ + j], part[j]);
#pragma unroll
  for (int j = 0; j < F_; ++j) sPart[tid * F_ + j] = part[j];
  __syncthreads();
  for (int s = 128; s > 0; s >>= 1) {
    if (tid < s)
#pragma unroll
      for (int j = 0; j < F_; ++j) sPart[tid * F_ + j] += sPart[(tid + s) * F_ + j];
    __syncthreads();
  }
  if (tid == 0) {
    float mx = -1e30f;
#pragma unroll
    for (int j = 0; j < F_; ++j) { float l = sPart[j] + bsel[j]; sSel4[j] = l; mx = fmaxf(mx, l); }
    float s = 0.f;
#pragma unroll
    for (int j = 0; j < F_; ++j) { float e = __expf(sSel4[j] - mx); sSel4[j] = e; s += e; }
    float inv = 1.f / s;
#pragma unroll
    for (int j = 0; j < F_; ++j) { sSel4[j] *= inv; selOut[b * F_ + j] = sSel4[j]; }
  }
  __syncthreads();
  // sel_weights [B,T,1,F] = tile(sel)
  for (int e = tid; e < T_ * F_; e += 256)
    selWOut[(size_t)b * T_ * F_ + e] = sSel4[e & (F_ - 1)];
}

// ---------------------------------------------------------------------------
// Fused GRN GEMM tile: M=64, N=512, K=512, bf16 WMMA with f32 accumulate.
// 8 waves; wave w owns output columns [64w, 64w+64) (4 N-tiles).
// Register blocking: 4 M-subtiles x 4 N-tiles = 16 acc tiles; each B fragment
// load is reused by 4 WMMAs (cuts L2 B-traffic 4x vs M=16).
// EPI 1: val = acc + bias      -> bf16 x2 out.
// EPI 2: g = sigmoid(acc+bias); gated = g*x2 + (1-g)*v -> bf16 out (x1 buffer).
// ---------------------------------------------------------------------------
template <int EPI>
__device__ __forceinline__ void gemm64(
    const unsigned short* sA,               // [64][512] bf16 bits (LDS)
    const unsigned short* __restrict__ Bt,  // [512(n)][512(k)] bf16 bits (global)
    const float* __restrict__ bias,         // [512]
    unsigned short* sOut,                   // [64][512] bf16 out (LDS)
    const unsigned short* sX2r,             // EPI2: x2 bf16 read (LDS)
    const float* sVv, int lane, int wave)
{
  const int r = lane & 15;
  const int half = lane >> 4;
  v8f acc[4][4];
#pragma unroll
  for (int m = 0; m < 4; ++m)
#pragma unroll
    for (int j = 0; j < 4; ++j)
      acc[m][j] = (v8f){0.f,0.f,0.f,0.f,0.f,0.f,0.f,0.f};

  for (int kb = 0; kb < H_; kb += 32) {
    // A fragments 16x32: lanes 0-15 K={kb..kb+7, kb+16..kb+23}, lanes 16-31 +8
    Frag16 a[4];
#pragma unroll
    for (int m = 0; m < 4; ++m) {
      const unsigned short* ap = sA + (m * 16 + r) * H_ + kb + half * 8;
      a[m].u[0] = *(const uint4*)(ap);
      a[m].u[1] = *(const uint4*)(ap + 16);
    }
#pragma unroll
    for (int j = 0; j < 4; ++j) {
      const int n0 = wave * 64 + j * 16;
      // B fragment 32x16: lane = col, lane-half 0 -> K={kb..kb+7, kb+16..kb+23}, half 1 -> +8
      Frag16 bfrag;
      const unsigned short* bp = Bt + (size_t)(n0 + r) * H_ + kb + half * 8;
      bfrag.u[0] = *(const uint4*)(bp);
      bfrag.u[1] = *(const uint4*)(bp + 16);
#pragma unroll
      for (int m = 0; m < 4; ++m)
        acc[m][j] = __builtin_amdgcn_wmma_f32_16x16x32_bf16(
            false, a[m].v, false, bfrag.v, (short)0, acc[m][j], false, false);
    }
  }
  // Epilogue: D element i of lane -> row = 16m + i + 8*half, col = n0 + (lane&15)
#pragma unroll
  for (int j = 0; j < 4; ++j) {
    const int n = wave * 64 + j * 16 + r;
    const float bv = bias[n];
#pragma unroll
    for (int m = 0; m < 4; ++m) {
#pragma unroll
      for (int i = 0; i < 8; ++i) {
        const int row = m * 16 + i + 8 * half;
        float val = acc[m][j][i] + bv;
        if (EPI == 1) {
          sOut[row * H_ + n] = f2bf(val);
        } else {
          float g   = sigmoidf_(val);
          float x2v = bf2f(sX2r[row * H_ + n]);
          sOut[row * H_ + n] = f2bf(g * x2v + (1.f - g) * sVv[row]);
        }
      }
    }
  }
}

// ---------------------------------------------------------------------------
// Kernel 3: fused per-variable GRNs + LN + sel-weighted sum over F.
// grid = BT_/MT_ = 512 workgroups x 256 threads (8 wave32s).
// LDS: x1/gated bf16 64KB + x2 bf16 64KB + acc f32 128KB = ~256.5KB (<320KB WGP).
// ---------------------------------------------------------------------------
__global__ __launch_bounds__(256) void vsn_main_kernel(
    const float* __restrict__ inputs,
    const float* __restrict__ w1, const float* __restrict__ b1,
    const unsigned short* __restrict__ W2t, const float* __restrict__ b2,
    const unsigned short* __restrict__ Wgt, const float* __restrict__ bg,
    const float* __restrict__ gamma, const float* __restrict__ beta,
    const float* __restrict__ sel, float* __restrict__ outSel)
{
  __shared__ __align__(16) unsigned short sX1[MT_ * H_];   // x1 then gated, bf16  64 KB
  __shared__ __align__(16) unsigned short sX2b[MT_ * H_];  // x2 bf16              64 KB
  __shared__ float sAcc[MT_ * H_];                          // selected f32        128 KB
  __shared__ float sV[MT_], sSel[MT_];

  const int tid  = threadIdx.x;
  const int lane = tid & 31;
  const int wave = tid >> 5;
  const int row0 = blockIdx.x * MT_;     // global row = b*T + t

  for (int i = tid; i < MT_ * H_; i += 256) sAcc[i] = 0.f;

  for (int f = 0; f < F_; ++f) {
    if (tid < MT_) {
      int row = row0 + tid;
      sV[tid]   = inputs[(size_t)row * F_ + f];          // inputs[b][t][f]
      sSel[tid] = sel[(row >> 7) * F_ + f];              // row>>7 == b (T=128)
    }
    __syncthreads();

    // x1 = elu(v * w1[f] + b1[f])  -> bf16 LDS
    for (int i = tid; i < MT_ * H_; i += 256) {
      int rr = i >> 9, h = i & (H_ - 1);
      float x = fmaf(sV[rr], w1[f * H_ + h], b1[f * H_ + h]);
      sX1[i] = f2bf(eluf(x));
    }
    __syncthreads();

    // x2 = x1 @ W2[f] + b2[f]  -> sX2b (bf16)
    gemm64<1>(sX1, W2t + (size_t)f * H_ * H_, b2 + f * H_, sX2b, nullptr, nullptr,
              lane, wave);
    __syncthreads();

    // gv = sigmoid(x2 @ Wg[f] + bg[f]); gated = gv*x2 + (1-gv)*v -> sX1 (bf16)
    gemm64<2>(sX2b, Wgt + (size_t)f * H_ * H_, bg + f * H_, sX1, sX2b, sV,
              lane, wave);
    __syncthreads();

    // LayerNorm per row + sel-weighted accumulate (8 rows per wave)
#pragma unroll
    for (int rr = 0; rr < 8; ++rr) {
      int row = wave * 8 + rr;
      float vals[16];
      float s1 = 0.f, s2 = 0.f;
#pragma unroll
      for (int i = 0; i < 16; ++i) {
        float x = bf2f(sX1[row * H_ + lane * 16 + i]);
        vals[i] = x; s1 += x; s2 += x * x;
      }
      for (int off = 16; off >= 1; off >>= 1) {
        s1 += __shfl_xor(s1, off, 32);
        s2 += __shfl_xor(s2, off, 32);
      }
      float m    = s1 * (1.f / H_);
      float var  = s2 * (1.f / H_) - m * m;
      float rstd = rsqrtf(var + EPS_);
      float w    = sSel[row];
#pragma unroll
      for (int i = 0; i < 16; ++i) {
        int h = lane * 16 + i;
        float tv = gamma[f * H_ + h] * (vals[i] - m) * rstd + beta[f * H_ + h];
        sAcc[row * H_ + h] += w * tv;
      }
    }
    __syncthreads();
  }

  for (int i = tid; i < MT_ * H_; i += 256) {
    int rr = i >> 9, h = i & (H_ - 1);
    outSel[(size_t)(row0 + rr) * H_ + h] = sAcc[i];
  }
}

// ---------------------------------------------------------------------------
extern "C" void kernel_launch(void* const* d_in, const int* in_sizes, int n_in,
                              void* d_out, int out_size, void* d_ws, size_t ws_size,
                              hipStream_t stream) {
  (void)in_sizes; (void)n_in; (void)out_size; (void)ws_size;

  const float* inputs  = (const float*)d_in[0];
  const float* W1c     = (const float*)d_in[1];
  const float* b1c     = (const float*)d_in[2];
  const float* W2c     = (const float*)d_in[3];
  const float* b2c     = (const float*)d_in[4];
  const float* Wgc     = (const float*)d_in[5];
  const float* bgc     = (const float*)d_in[6];
  const float* gamma_c = (const float*)d_in[7];
  const float* beta_c  = (const float*)d_in[8];
  const float* Wsel    = (const float*)d_in[9];
  const float* bsel    = (const float*)d_in[10];
  const float* w1      = (const float*)d_in[11];
  const float* b1      = (const float*)d_in[12];
  const float* W2      = (const float*)d_in[13];
  const float* b2      = (const float*)d_in[14];
  const float* Wg      = (const float*)d_in[15];
  const float* bg      = (const float*)d_in[16];
  const float* gamma   = (const float*)d_in[17];
  const float* beta    = (const float*)d_in[18];

  // workspace: [W2t bf16 2MB][Wgt bf16 2MB][sel f32 4KB]
  unsigned short* W2t = (unsigned short*)d_ws;
  unsigned short* Wgt = (unsigned short*)((char*)d_ws + (size_t)F_ * H_ * H_ * 2);
  float*          selW = (float*)((char*)d_ws + (size_t)F_ * H_ * H_ * 4);

  float* outSel  = (float*)d_out;                   // selected [B,T,H]
  float* outSelW = outSel + (size_t)B_ * T_ * H_;   // sel_weights [B,T,1,F]

  vsn_prep_kernel<<<(F_ * H_ * H_) / 256, 256, 0, stream>>>(W2, Wg, W2t, Wgt);
  vsn_context_kernel<<<B_, 256, 0, stream>>>(inputs, W1c, b1c, W2c, b2c, Wgc, bgc,
                                             gamma_c, beta_c, Wsel, bsel, selW, outSelW);
  vsn_main_kernel<<<BT_ / MT_, 256, 0, stream>>>(inputs, w1, b1, W2t, b2, Wgt, bg,
                                                 gamma, beta, selW, outSel);
}
